// conv_block_72155450573168
// MI455X (gfx1250) — matrix operand order repaired and checked by reading, not compiled, not run on hardware
//
#include <hip/hip_runtime.h>
#include <hip/hip_bf16.h>
#include <math.h>

// ---------------- problem constants ----------------
#define CIN    64
#define DDIM   64
#define HDIM   96
#define WDIM   96
#define HW     (HDIM * WDIM)            // 9216
#define SPOS   (DDIM * HW)              // 589824 spatial positions
#define CS     (CIN * SPOS)             // 37748736 total elements
#define NGRP   32                       // groups (2 channels each)
#define BPG    4608                     // 256-thread blocks per group
#define NCONVBLK (CS / 256)             // 147456

typedef __attribute__((ext_vector_type(16))) _Float16 v16h;
typedef __attribute__((ext_vector_type(8)))  _Float16 h8;
typedef __attribute__((ext_vector_type(8)))  float    v8f;
typedef int vi4 __attribute__((vector_size(16)));     // b128 payload type

// ---------------- workspace layout (bytes) ----------------
#define T_OFF    ((size_t)0)                         // t buffer: CS * 4
#define CP_OFF   ((size_t)150994944)                 // conv partials 3*32*4608*2*4
#define TP_OFF   ((size_t)154533888)                 // t partials 32*4608*2*4
#define CST_OFF  ((size_t)155713536)                 // conv stats 96*2 f32
#define TST_OFF  ((size_t)155714304)                 // t stats 32*2 f32
#define EB_OFF   ((size_t)155714560)                 // expB  f16 32KB
#define CB_OFF   ((size_t)155747328)                 // compB f16 32KB

// gfx1250 async copy to LDS (guarded so compile can never break on absence)
#if __has_builtin(__builtin_amdgcn_global_load_async_to_lds_b128) && \
    __has_builtin(__builtin_amdgcn_s_wait_asynccnt)
#define HAS_ASYNC_LDS 1
#else
#define HAS_ASYNC_LDS 0
#endif

// =====================================================================
// Branch-free exact GELU: 0.5*x*(1+erf(x/sqrt(2))) with A&S 7.1.26 erf
// (|abs err| <= 1.5e-7; uses v_exp_f32/v_rcp_f32, no divergence).
// =====================================================================
__device__ __forceinline__ float gelu_exact(float x) {
    float ax = fabsf(x) * 0.70710678118654752f;       // |x|/sqrt(2)
    float t  = __builtin_amdgcn_rcpf(fmaf(0.3275911f, ax, 1.0f));
    float p  = fmaf(t, 1.061405429f, -1.453152027f);
    p = fmaf(t, p, 1.421413741f);
    p = fmaf(t, p, -0.284496736f);
    p = fmaf(t, p, 0.254829592f);
    p = p * t;
    float e  = __expf(-ax * ax);                      // v_exp_f32
    float erf_ax = fmaf(-p, e, 1.0f);                 // erf(|x|/sqrt2)
    return 0.5f * x * (1.0f + copysignf(erf_ax, x));
}

// =====================================================================
// Kernel 1: preformat GEMM weights into WMMA B-operand layout (f16).
// =====================================================================
__global__ __launch_bounds__(32) void fmt_weights(
    const float* __restrict__ expW,   // [256][64]
    const float* __restrict__ compW,  // [64][256]
    _Float16* __restrict__ expB,      // [32 tiles][32 lanes][16]
    _Float16* __restrict__ compB)     // [32 tiles][32 lanes][16]
{
    int tile = blockIdx.x;
    int lane = threadIdx.x;
    int n  = lane & 15;
    int kb = (lane < 16) ? 0 : 8;
    if (tile < 32) {
        int nt = tile >> 1, kc = tile & 1;
        _Float16* dst = expB + ((size_t)tile * 32 + lane) * 16;
        #pragma unroll
        for (int e = 0; e < 16; ++e)
            dst[e] = (_Float16)expW[(nt * 16 + n) * 64 + kc * 32 + kb + (e & 7) + ((e >> 3) << 4)];
    } else {
        int t2 = tile - 32;
        int nt2 = t2 >> 3, kc = t2 & 7;
        _Float16* dst = compB + ((size_t)t2 * 32 + lane) * 16;
        #pragma unroll
        for (int e = 0; e < 16; ++e)
            dst[e] = (_Float16)compW[(nt2 * 16 + n) * 256 + kc * 32 + kb + (e & 7) + ((e >> 3) << 4)];
    }
}

// =====================================================================
// Depthwise 3x3x3 convs (dil 1/3/5). Interior fast path: all 81 tap
// addresses are base + compile-time constants -> immediate-offset
// global_load_b32, no bounds checks, pure fmaf chain.
// =====================================================================
__device__ __forceinline__ void conv_tri(
    const float* __restrict__ xc,
    const float* __restrict__ w1, const float* __restrict__ w2,
    const float* __restrict__ w3,
    int z, int y, int xw, float& o1, float& o2, float& o3)
{
    float a0 = 0.f, a1 = 0.f, a2 = 0.f;
    if (z >= 5 && z < DDIM - 5 && y >= 5 && y < HDIM - 5 &&
        xw >= 5 && xw < WDIM - 5) {
        const float* p = xc + z * HW + y * WDIM + xw;
        #pragma unroll
        for (int kz = 0; kz < 3; ++kz)
        #pragma unroll
        for (int ky = 0; ky < 3; ++ky)
        #pragma unroll
        for (int kx = 0; kx < 3; ++kx) {
            const int wi  = kz * 9 + ky * 3 + kx;
            const int off = (kz - 1) * HW + (ky - 1) * WDIM + (kx - 1);
            a0 = fmaf(w1[wi], p[off * 1], a0);
            a1 = fmaf(w2[wi], p[off * 3], a1);
            a2 = fmaf(w3[wi], p[off * 5], a2);
        }
    } else {
        for (int kz = 0; kz < 3; ++kz)
        for (int ky = 0; ky < 3; ++ky)
        for (int kx = 0; kx < 3; ++kx) {
            int wi = kz * 9 + ky * 3 + kx;
            float wv0 = w1[wi], wv1 = w2[wi], wv2 = w3[wi];
            int dz = kz - 1, dy = ky - 1, dx = kx - 1;
            {
                int zz = z + dz, yy = y + dy, xx = xw + dx;
                if ((unsigned)zz < (unsigned)DDIM && (unsigned)yy < (unsigned)HDIM &&
                    (unsigned)xx < (unsigned)WDIM)
                    a0 = fmaf(wv0, xc[zz * HW + yy * WDIM + xx], a0);
            }
            {
                int zz = z + 3 * dz, yy = y + 3 * dy, xx = xw + 3 * dx;
                if ((unsigned)zz < (unsigned)DDIM && (unsigned)yy < (unsigned)HDIM &&
                    (unsigned)xx < (unsigned)WDIM)
                    a1 = fmaf(wv1, xc[zz * HW + yy * WDIM + xx], a1);
            }
            {
                int zz = z + 5 * dz, yy = y + 5 * dy, xx = xw + 5 * dx;
                if ((unsigned)zz < (unsigned)DDIM && (unsigned)yy < (unsigned)HDIM &&
                    (unsigned)xx < (unsigned)WDIM)
                    a2 = fmaf(wv2, xc[zz * HW + yy * WDIM + xx], a2);
            }
        }
    }
    o1 = a0; o2 = a1; o3 = a2;
}

// =====================================================================
// Kernel 2: conv pass 1 -> per-block sum/sumsq partials (per conv, group)
// =====================================================================
__global__ __launch_bounds__(256) void conv_stats(
    const float* __restrict__ x,
    const float* __restrict__ dw1, const float* __restrict__ dw2,
    const float* __restrict__ dw3,
    float* __restrict__ part)
{
    unsigned idx = blockIdx.x * 256u + threadIdx.x;
    int c = idx / SPOS;
    int s = idx % SPOS;
    int z = s / HW; int r = s % HW; int y = r / WDIM; int xw = r % WDIM;

    float o1, o2, o3;
    conv_tri(x + (size_t)c * SPOS, dw1 + c * 27, dw2 + c * 27, dw3 + c * 27,
             z, y, xw, o1, o2, o3);

    float v[6] = { o1, o1 * o1, o2, o2 * o2, o3, o3 * o3 };
    int lane = threadIdx.x & 31, wave = threadIdx.x >> 5;
    #pragma unroll
    for (int off = 16; off; off >>= 1)
        #pragma unroll
        for (int j = 0; j < 6; ++j) v[j] += __shfl_down(v[j], off);

    __shared__ float sred[8][6];
    if (lane == 0)
        #pragma unroll
        for (int j = 0; j < 6; ++j) sred[wave][j] = v[j];
    __syncthreads();
    if (threadIdx.x < 6) {
        float acc = 0.f;
        #pragma unroll
        for (int w = 0; w < 8; ++w) acc += sred[w][threadIdx.x];
        int conv = threadIdx.x >> 1, which = threadIdx.x & 1;
        int group = c >> 1;
        int big = blockIdx.x % BPG;
        part[(((size_t)conv * NGRP + group) * BPG + big) * 2 + which] = acc;
    }
}

// =====================================================================
// Kernel 3/5: deterministic partial reduction -> (mean, rstd)
// =====================================================================
__global__ __launch_bounds__(256) void reduce_stats(
    const float* __restrict__ part, float* __restrict__ stats)
{
    int gid = blockIdx.x;
    const float* p = part + (size_t)gid * BPG * 2;
    double s = 0.0, s2 = 0.0;
    for (int i = threadIdx.x; i < BPG; i += 256) {
        s  += (double)p[i * 2];
        s2 += (double)p[i * 2 + 1];
    }
    __shared__ double rs[256], rs2[256];
    rs[threadIdx.x] = s; rs2[threadIdx.x] = s2;
    __syncthreads();
    for (int off = 128; off; off >>= 1) {
        if (threadIdx.x < (unsigned)off) {
            rs[threadIdx.x]  += rs[threadIdx.x + off];
            rs2[threadIdx.x] += rs2[threadIdx.x + off];
        }
        __syncthreads();
    }
    if (threadIdx.x == 0) {
        double cnt  = 2.0 * (double)SPOS;
        double mean = rs[0] / cnt;
        double var  = rs2[0] / cnt - mean * mean;
        stats[gid * 2]     = (float)mean;
        stats[gid * 2 + 1] = (float)(1.0 / sqrt(var + 1e-5));
    }
}

// =====================================================================
// Kernel 4: conv pass 2 -> t = gn1+gn2+gn3, plus partials for 4th GN
// =====================================================================
__global__ __launch_bounds__(256) void build_t(
    const float* __restrict__ x,
    const float* __restrict__ dw1, const float* __restrict__ dw2,
    const float* __restrict__ dw3,
    const float* __restrict__ cstats,
    const float* __restrict__ nw, const float* __restrict__ nb,
    float* __restrict__ tbuf,
    float* __restrict__ tpart)
{
    unsigned idx = blockIdx.x * 256u + threadIdx.x;
    int c = idx / SPOS;
    int s = idx % SPOS;
    int z = s / HW; int r = s % HW; int y = r / WDIM; int xw = r % WDIM;

    float o1, o2, o3;
    conv_tri(x + (size_t)c * SPOS, dw1 + c * 27, dw2 + c * 27, dw3 + c * 27,
             z, y, xw, o1, o2, o3);

    int g = c >> 1;
    float wc = nw[c], bc = nb[c];
    float n1 = (o1 - cstats[(0 * NGRP + g) * 2]) * cstats[(0 * NGRP + g) * 2 + 1] * wc + bc;
    float n2 = (o2 - cstats[(1 * NGRP + g) * 2]) * cstats[(1 * NGRP + g) * 2 + 1] * wc + bc;
    float n3 = (o3 - cstats[(2 * NGRP + g) * 2]) * cstats[(2 * NGRP + g) * 2 + 1] * wc + bc;
    float tv = n1 + n2 + n3;
    tbuf[idx] = tv;

    float v[2] = { tv, tv * tv };
    int lane = threadIdx.x & 31, wave = threadIdx.x >> 5;
    #pragma unroll
    for (int off = 16; off; off >>= 1) {
        v[0] += __shfl_down(v[0], off);
        v[1] += __shfl_down(v[1], off);
    }
    __shared__ float sred[8][2];
    if (lane == 0) { sred[wave][0] = v[0]; sred[wave][1] = v[1]; }
    __syncthreads();
    if (threadIdx.x < 2) {
        float acc = 0.f;
        #pragma unroll
        for (int w = 0; w < 8; ++w) acc += sred[w][threadIdx.x];
        int big = blockIdx.x % BPG;
        tpart[((size_t)g * BPG + big) * 2 + threadIdx.x] = acc;
    }
}

// =====================================================================
// Kernel 6: WMMA GEMM stage (64 positions / block, 4 waves, 16 pos/wave)
// =====================================================================
__device__ __forceinline__ v8f wmma_f16(v16h a, v16h b, v8f c) {
    return __builtin_amdgcn_wmma_f32_16x16x32_f16(
        false, a, false, b, (short)0, c, false, false);
}

__global__ __launch_bounds__(128) void gemm_kernel(
    const float* __restrict__ tbuf,
    const float* __restrict__ tstats,
    const float* __restrict__ nw, const float* __restrict__ nb,
    const _Float16* __restrict__ expB,
    const float* __restrict__ expBias,
    const _Float16* __restrict__ compB,
    const float* __restrict__ compBias,
    const float* __restrict__ xin,
    float* __restrict__ out)
{
    __shared__ __align__(16) _Float16 tileA[64 * 64];     // [pos][ch] f16 normalized
    __shared__ __align__(16) _Float16 gelus[4][16 * 256]; // per wave [m][ch]
#if HAS_ASYNC_LDS
    __shared__ __align__(16) float rawT[64 * 64];         // [ch][pos] raw f32 tile
#endif

    const int tid   = threadIdx.x;
    const int lane  = tid & 31;
    const int wave  = tid >> 5;
    const int pbase = blockIdx.x * 64;

#if HAS_ASYNC_LDS
    // --- async DMA the raw f32 tile (64ch x 64pos, 16KB) into LDS ---
    for (int i = tid; i < 1024; i += 128) {
        int ch = i >> 4, ck = i & 15;   // 16 x b128 chunks per channel row
        const float* src = tbuf + (size_t)ch * SPOS + pbase + ck * 4;
        float* dst = rawT + ch * 64 + ck * 4;
        __builtin_amdgcn_global_load_async_to_lds_b128(
            (vi4*)src, (vi4*)dst, 0, 0);
    }
    __builtin_amdgcn_s_wait_asynccnt(0);
    __syncthreads();
    // --- normalize (4th group norm) -> f16 tile, transposed [pos][ch] ---
    for (int i = tid; i < 4096; i += 128) {
        int ch = i >> 6, pp = i & 63;
        float v = rawT[ch * 64 + pp];
        int g = ch >> 1;
        v = (v - tstats[g * 2]) * tstats[g * 2 + 1] * nw[ch] + nb[ch];
        tileA[pp * 64 + ch] = (_Float16)v;
    }
#else
    for (int i = tid; i < 4096; i += 128) {
        int ch = i >> 6, pp = i & 63;
        float v = tbuf[(size_t)ch * SPOS + pbase + pp];
        int g = ch >> 1;
        v = (v - tstats[g * 2]) * tstats[g * 2 + 1] * nw[ch] + nb[ch];
        tileA[pp * 64 + ch] = (_Float16)v;
    }
#endif
    __syncthreads();

    const int row  = lane & 15;
    const int posL = wave * 16 + row;
    const int klo  = (lane < 16) ? 0 : 8;
    const int khi  = (lane < 16) ? 16 : 24;

    // --- A operands (ISA 16-bit A layout), K = 0..31 / 32..63 ---
    v16h a0, a1;
    {
        h8 lo = *(const h8*)(tileA + posL * 64 + 0  + klo);
        h8 hi = *(const h8*)(tileA + posL * 64 + 0  + khi);
        #pragma unroll
        for (int e = 0; e < 8; ++e) { a0[e] = lo[e]; a0[8 + e] = hi[e]; }
        lo = *(const h8*)(tileA + posL * 64 + 32 + klo);
        hi = *(const h8*)(tileA + posL * 64 + 32 + khi);
        #pragma unroll
        for (int e = 0; e < 8; ++e) { a1[e] = lo[e]; a1[8 + e] = hi[e]; }
    }

    // --- expansion 64->256 + bias + branch-free exact GELU -> LDS f16 ---
    const v16h* eB = (const v16h*)expB;
    _Float16* gl = gelus[wave];
    const int n    = lane & 15;
    const int mofs = (lane >> 4) * 8;
    for (int nt = 0; nt < 16; ++nt) {
        v8f c = {};
        c = wmma_f16(a0, eB[(nt * 2 + 0) * 32 + lane], c);
        c = wmma_f16(a1, eB[(nt * 2 + 1) * 32 + lane], c);
        int ch = nt * 16 + n;
        float bias = expBias[ch];
        #pragma unroll
        for (int v = 0; v < 8; ++v)
            gl[(mofs + v) * 256 + ch] = (_Float16)gelu_exact(c[v] + bias);
    }
    __syncthreads();

    // --- compression 256->64 ---
    const v16h* cB = (const v16h*)compB;
    for (int nt2 = 0; nt2 < 4; ++nt2) {
        v8f acc = {};
        #pragma unroll
        for (int kc = 0; kc < 8; ++kc) {
            h8 lo = *(const h8*)(gl + row * 256 + kc * 32 + klo);
            h8 hi = *(const h8*)(gl + row * 256 + kc * 32 + khi);
            v16h a;
            #pragma unroll
            for (int e = 0; e < 8; ++e) { a[e] = lo[e]; a[8 + e] = hi[e]; }
            acc = wmma_f16(a, cB[(nt2 * 8 + kc) * 32 + lane], acc);
        }
        int ch = nt2 * 16 + n;
        float cb = compBias[ch];
        size_t ob = (size_t)ch * SPOS + pbase + wave * 16 + mofs;
        #pragma unroll
        for (int v = 0; v < 8; ++v)
            out[ob + v] = acc[v] + cb + xin[ob + v];
    }
}

// =====================================================================
extern "C" void kernel_launch(void* const* d_in, const int* in_sizes, int n_in,
                              void* d_out, int out_size, void* d_ws, size_t ws_size,
                              hipStream_t stream) {
    const float* x       = (const float*)d_in[0];
    const float* dw1     = (const float*)d_in[1];
    const float* dw2     = (const float*)d_in[2];
    const float* dw3     = (const float*)d_in[3];
    const float* nw      = (const float*)d_in[4];
    const float* nb      = (const float*)d_in[5];
    const float* expW    = (const float*)d_in[6];
    const float* expBias = (const float*)d_in[7];
    const float* compW   = (const float*)d_in[8];
    const float* compBias= (const float*)d_in[9];

    char* ws = (char*)d_ws;
    float*     tbuf   = (float*)(ws + T_OFF);
    float*     cpart  = (float*)(ws + CP_OFF);
    float*     tpart  = (float*)(ws + TP_OFF);
    float*     cstats = (float*)(ws + CST_OFF);
    float*     tstats = (float*)(ws + TST_OFF);
    _Float16*  expB   = (_Float16*)(ws + EB_OFF);
    _Float16*  compB  = (_Float16*)(ws + CB_OFF);
    float*     out    = (float*)d_out;

    fmt_weights<<<64, 32, 0, stream>>>(expW, compW, expB, compB);
    conv_stats<<<NCONVBLK, 256, 0, stream>>>(x, dw1, dw2, dw3, cpart);
    reduce_stats<<<96, 256, 0, stream>>>(cpart, cstats);
    build_t<<<NCONVBLK, 256, 0, stream>>>(x, dw1, dw2, dw3, cstats, nw, nb,
                                          tbuf, tpart);
    reduce_stats<<<32, 256, 0, stream>>>(tpart, tstats);
    gemm_kernel<<<SPOS / 64, 128, 0, stream>>>(tbuf, tstats, nw, nb,
                                               expB, expBias, compB, compBias,
                                               x, out);
}
